// LearnableAlign_24653112279443
// MI455X (gfx1250) — matrix operand order repaired and checked by reading, not compiled, not run on hardware
//
#include <hip/hip_runtime.h>
#include <hip/hip_bf16.h>

typedef __bf16 v16bf __attribute__((ext_vector_type(16)));
typedef float  v8f   __attribute__((ext_vector_type(8)));

#define CM        128
#define N_TOK     65536
#define TILE_TOK  64
#define LSTR      136   // 128 + 8 pad elements; row = 272B, 16B aligned
#define WMAT      16384 // ushorts per swizzled 128x128 bf16 matrix

__device__ __forceinline__ unsigned short f2bf(float f) {
  unsigned int u = __float_as_uint(f);
  u += 0x7fffu + ((u >> 16) & 1u);           // round-to-nearest-even
  return (unsigned short)(u >> 16);
}

__device__ __forceinline__ v16bf make_v16bf(uint4 a, uint4 b) {
  union { uint4 q[2]; v16bf v; } u;
  u.q[0] = a; u.q[1] = b;
  return u.v;
}

// Swizzled B-fragment layout (per 32x16 frag = 512 bf16):
//   [0..255]   : lanes' first 8 elems  (lane L at L*8)   -> ds_load_b128, lane L at dword 4L
//   [256..511] : lanes' second 8 elems (lane L at 256+L*8)
// Bank-optimal: lanes 0-15 hit banks 0..63 once, lanes 16-31 the next cycle.
__device__ __forceinline__ v16bf load_bfrag(const unsigned short* sw,
                                            int et, int kt, int lane) {
  const unsigned short* base = sw + (((et << 2) + kt) << 9);
  return make_v16bf(*(const uint4*)(base + lane * 8),
                    *(const uint4*)(base + 256 + lane * 8));
}

#define WMMA_BF16(A, B, C) \
  __builtin_amdgcn_wmma_f32_16x16x32_bf16(false, (A), false, (B), (short)0, (C), false, false)

// ---------------------------------------------------------------------------
// Prep 1: W3 = (fuse_w2 @ out_w) @ v_w ; c2 = (fuse_w2@out_w)@v_b ;
//         btot = fuse_w2@out_b + fuse_b.   One block per output row o2.
// ---------------------------------------------------------------------------
__global__ __launch_bounds__(128) void prep_w3_kernel(
    const float* __restrict__ v_w,   const float* __restrict__ v_b,
    const float* __restrict__ out_w, const float* __restrict__ out_b,
    const float* __restrict__ fuse_w, const float* __restrict__ fuse_b,
    float* __restrict__ w3_tmp, float* __restrict__ c2, float* __restrict__ btot)
{
  __shared__ float fw2row[CM];
  __shared__ float w2row[CM];
  const int o2 = blockIdx.x;
  const int t  = threadIdx.x;
  fw2row[t] = fuse_w[o2 * 2 * CM + CM + t];
  __syncthreads();
  float acc = 0.f;
  for (int o = 0; o < CM; ++o) acc += fw2row[o] * out_w[o * CM + t];   // W2[o2][t]
  w2row[t] = acc;
  __syncthreads();
  float a3 = 0.f;
  for (int e = 0; e < CM; ++e) a3 += w2row[e] * v_w[e * CM + t];       // W3[o2][t]
  w3_tmp[o2 * CM + t] = a3;
  if (t == 0) {
    float s = 0.f;
    for (int e = 0; e < CM; ++e) s += w2row[e] * v_b[e];
    c2[o2] = s;
    float s2 = 0.f;
    for (int o = 0; o < CM; ++o) s2 += fw2row[o] * out_b[o];
    btot[o2] = s2 + fuse_b[o2];
  }
}

// ---------------------------------------------------------------------------
// Prep 2: convert q_w, k_w, fuse_w1, W3 to bf16 in split-half B-fragment order.
// B (32x16, 16-bit): lane L = column N = L%16, elem j -> K = 8*(L/16)+(j&7)+16*(j>>3).
// Dest index within a matrix: frag*512 + half*256 + L*8 + (j&7), half = j>>3.
// ---------------------------------------------------------------------------
__global__ __launch_bounds__(256) void prep_swz_kernel(
    const float* __restrict__ q_w, const float* __restrict__ k_w,
    const float* __restrict__ fuse_w, const float* __restrict__ w3_tmp,
    unsigned short* __restrict__ dst)
{
  const int idx = blockIdx.x * 256 + threadIdx.x;   // 4 * 16384
  const int mat  = idx >> 14;
  const int r    = idx & 16383;
  const int jj   = r & 7;
  const int L    = (r >> 3) & 31;
  const int half = (r >> 8) & 1;
  const int kt   = (r >> 9) & 3;
  const int et   = r >> 11;
  const int j    = half * 8 + jj;
  const int e    = et * 16 + (L & 15);
  const int col  = kt * 32 + ((L >> 4) << 3) + (half << 4) + jj;
  float v;
  if      (mat == 0) v = q_w[e * CM + col];
  else if (mat == 1) v = k_w[e * CM + col];
  else if (mat == 2) v = fuse_w[e * 2 * CM + col];   // fuse_w1: cols [0,128)
  else               v = w3_tmp[e * CM + col];
  dst[idx] = f2bf(v);
}

// ---------------------------------------------------------------------------
// Stage a 64-token tile of both modalities, [chan][tok] global -> [tok][chan]
// bf16 in LDS (transpose + convert). 128 threads, float4 coalesced loads.
// ---------------------------------------------------------------------------
__device__ __forceinline__ void stage_tile(const float* __restrict__ x,
                                           int b, int n0, int tid,
                                           unsigned short* m1t, unsigned short* m2t)
{
  for (int it = 0; it < 32; ++it) {
    const int c  = it * 8 + (tid >> 4);
    const int jj = (tid & 15) * 4;
    const float4 v = *(const float4*)(x + ((size_t)(b * 2 * CM + c)) * N_TOK + n0 + jj);
    unsigned short* dst = (c < CM) ? m1t : m2t;
    const int cc = c & (CM - 1);
    dst[(jj + 0) * LSTR + cc] = f2bf(v.x);
    dst[(jj + 1) * LSTR + cc] = f2bf(v.y);
    dst[(jj + 2) * LSTR + cc] = f2bf(v.z);
    dst[(jj + 3) * LSTR + cc] = f2bf(v.w);
  }
}

// Copy one swizzled bf16 weight matrix pair (2*16384 ushorts) global -> LDS.
__device__ __forceinline__ void stage_weights(const unsigned short* __restrict__ g,
                                              unsigned short* l, int tid)
{
  const uint4* src = (const uint4*)g;
  uint4* dst = (uint4*)l;
  for (int i = tid; i < 4096; i += 128) dst[i] = src[i];
}

// A-fragments (16x32 bf16): lane L holds row M=L%16; elems 0-7 -> K=8h+j,
// elems 8-15 -> K=16+8h+j, h=L/16. Both chunks are aligned 16B LDS reads.
__device__ __forceinline__ v16bf load_afrag(const unsigned short* tile,
                                            int row, int kt, int h)
{
  const unsigned short* p = tile + row * LSTR + kt * 32 + 8 * h;
  return make_v16bf(*(const uint4*)p, *(const uint4*)(p + 16));
}

// ---------------------------------------------------------------------------
// Pass 1: affinity[b][n] = (m1@q_w^T + q_b) . (m2@k_w^T + k_b) / sqrt(CM)
// ---------------------------------------------------------------------------
__global__ __launch_bounds__(128) void aff_kernel(
    const float* __restrict__ x,
    const unsigned short* __restrict__ qkw_sw,   // q at 0, k at +WMAT
    const float* __restrict__ q_b, const float* __restrict__ k_b,
    float* __restrict__ aff)
{
  __shared__ __align__(16) unsigned short ldsw[2 * WMAT];          // 64 KB
  __shared__ __align__(16) unsigned short tiles[2 * TILE_TOK * LSTR];
  __shared__ float qbl[CM];
  __shared__ float kbl[CM];
  unsigned short* m1t = tiles;
  unsigned short* m2t = tiles + TILE_TOK * LSTR;

  const int b   = blockIdx.x >> 10;
  const int n0  = (blockIdx.x & 1023) << 6;
  const int tid = threadIdx.x;

  stage_tile(x, b, n0, tid, m1t, m2t);
  stage_weights(qkw_sw, ldsw, tid);
  qbl[tid] = q_b[tid];
  kbl[tid] = k_b[tid];
  __syncthreads();

  const int lane = tid & 31;
  const int wv   = tid >> 5;
  const int m    = lane & 15;
  const int h    = lane >> 4;
  const int row  = wv * 16 + m;

  v16bf a1[4], a2[4];
  for (int kt = 0; kt < 4; ++kt) {
    a1[kt] = load_afrag(m1t, row, kt, h);
    a2[kt] = load_afrag(m2t, row, kt, h);
  }

  const unsigned short* qws = ldsw;
  const unsigned short* kws = ldsw + WMAT;

  float part[8] = {0.f, 0.f, 0.f, 0.f, 0.f, 0.f, 0.f, 0.f};
  for (int et = 0; et < 8; ++et) {
    // load all 8 B fragments for this etile first (one ds clause, one wait),
    // then run the two independent WMMA chains interleaved.
    v16bf bq[4], bk[4];
    for (int kt = 0; kt < 4; ++kt) bq[kt] = load_bfrag(qws, et, kt, lane);
    for (int kt = 0; kt < 4; ++kt) bk[kt] = load_bfrag(kws, et, kt, lane);
    v8f cq = {0.f, 0.f, 0.f, 0.f, 0.f, 0.f, 0.f, 0.f};
    v8f ck = {0.f, 0.f, 0.f, 0.f, 0.f, 0.f, 0.f, 0.f};
    for (int kt = 0; kt < 4; ++kt) {
      cq = WMMA_BF16(a1[kt], bq[kt], cq);
      ck = WMMA_BF16(a2[kt], bk[kt], ck);
    }
    const float qb = qbl[et * 16 + m];   // column N = lane%16
    const float kb = kbl[et * 16 + m];
    for (int r = 0; r < 8; ++r) part[r] += (cq[r] + qb) * (ck[r] + kb);
  }
  // reduce over N (16 lanes of each half-wave); C/D: lane holds M=r+8h, N=lane%16
  for (int s = 1; s <= 8; s <<= 1)
    for (int r = 0; r < 8; ++r)
      part[r] += __shfl_xor(part[r], s, 32);
  if (m == 0) {
    const float sc = 0.08838834764831845f;   // 1/sqrt(128)
    for (int r = 0; r < 8; ++r)
      aff[(size_t)b * N_TOK + n0 + wv * 16 + r + 8 * h] = part[r] * sc;
  }
}

// ---------------------------------------------------------------------------
// Pass 2: per-batch softmax stats (max, 1/sum of exp)
// ---------------------------------------------------------------------------
__global__ __launch_bounds__(256) void softmax_stats_kernel(
    const float* __restrict__ aff, float* __restrict__ stats)
{
  __shared__ float red[256];
  const int b = blockIdx.x, tid = threadIdx.x;
  const float* a = aff + (size_t)b * N_TOK;

  float mx = -3.4e38f;
  for (int t = tid; t < N_TOK; t += 256) mx = fmaxf(mx, a[t]);
  red[tid] = mx; __syncthreads();
  for (int s = 128; s > 0; s >>= 1) {
    if (tid < s) red[tid] = fmaxf(red[tid], red[tid + s]);
    __syncthreads();
  }
  mx = red[0]; __syncthreads();

  float sm = 0.f;
  for (int t = tid; t < N_TOK; t += 256) sm += __expf(a[t] - mx);
  red[tid] = sm; __syncthreads();
  for (int s = 128; s > 0; s >>= 1) {
    if (tid < s) red[tid] += red[tid + s];
    __syncthreads();
  }
  if (tid == 0) { stats[b * 2] = mx; stats[b * 2 + 1] = 1.0f / red[0]; }
}

// ---------------------------------------------------------------------------
// Pass 3: fused[n] = m1@fw1^T + w[n]*(m2@W3^T + c2) + btot ; transpose store.
// ---------------------------------------------------------------------------
__global__ __launch_bounds__(128) void out_kernel(
    const float* __restrict__ x,
    const unsigned short* __restrict__ fw_sw,    // fw1 at 0, W3 at +WMAT
    const float* __restrict__ c2, const float* __restrict__ btot,
    const float* __restrict__ aff, const float* __restrict__ stats,
    float* __restrict__ out)
{
  __shared__ __align__(16) unsigned short ldsw[2 * WMAT];          // 64 KB
  __shared__ __align__(16) unsigned short tiles[2 * TILE_TOK * LSTR];
  __shared__ float affl[TILE_TOK];
  __shared__ float c2l[CM];
  __shared__ float btl[CM];
  unsigned short* m1t = tiles;
  unsigned short* m2t = tiles + TILE_TOK * LSTR;
  float* outb = (float*)tiles;                 // reuse: [64][130] f32, 33280B

  const int b   = blockIdx.x >> 10;
  const int n0  = (blockIdx.x & 1023) << 6;
  const int tid = threadIdx.x;

  stage_tile(x, b, n0, tid, m1t, m2t);
  stage_weights(fw_sw, ldsw, tid);
  if (tid < TILE_TOK) affl[tid] = aff[(size_t)b * N_TOK + n0 + tid];
  c2l[tid] = c2[tid];
  btl[tid] = btot[tid];
  __syncthreads();

  const float mx  = stats[b * 2];
  const float inv = stats[b * 2 + 1];

  const int lane = tid & 31;
  const int wv   = tid >> 5;
  const int m    = lane & 15;
  const int h    = lane >> 4;
  const int row  = wv * 16 + m;

  v16bf a1[4], a2[4];
  for (int kt = 0; kt < 4; ++kt) {
    a1[kt] = load_afrag(m1t, row, kt, h);
    a2[kt] = load_afrag(m2t, row, kt, h);
  }
  float w8[8];
  for (int r = 0; r < 8; ++r)
    w8[r] = __expf(affl[wv * 16 + r + 8 * h] - mx) * inv;   // softmax weight of row M
  __syncthreads();   // all tile reads done before outb (aliased LDS) is written

  const unsigned short* fws = ldsw;
  const unsigned short* w3s = ldsw + WMAT;

  float res[8][8];
  for (int et = 0; et < 8; ++et) {
    v16bf bf[4], bs[4];
    for (int kt = 0; kt < 4; ++kt) bf[kt] = load_bfrag(fws, et, kt, lane);
    for (int kt = 0; kt < 4; ++kt) bs[kt] = load_bfrag(w3s, et, kt, lane);
    v8f f = {0.f, 0.f, 0.f, 0.f, 0.f, 0.f, 0.f, 0.f};
    v8f s = {0.f, 0.f, 0.f, 0.f, 0.f, 0.f, 0.f, 0.f};
    for (int kt = 0; kt < 4; ++kt) {
      f = WMMA_BF16(a1[kt], bf[kt], f);
      s = WMMA_BF16(a2[kt], bs[kt], s);
    }
    const float c2v = c2l[et * 16 + m];
    const float btv = btl[et * 16 + m];
    for (int r = 0; r < 8; ++r)
      res[et][r] = f[r] + w8[r] * (s[r] + c2v) + btv;
  }
  for (int et = 0; et < 8; ++et)
    for (int r = 0; r < 8; ++r)
      outb[(wv * 16 + r + 8 * h) * 130 + et * 16 + m] = res[et][r];
  __syncthreads();

  // coalesced channel-major store: out[b][e][n0+j]
  for (int e2 = 0; e2 < 64; ++e2) {
    const int e = e2 * 2 + (tid >> 6);
    const int j = tid & 63;
    out[((size_t)(b * CM + e)) * N_TOK + n0 + j] = outb[j * 130 + e];
  }
}

// ---------------------------------------------------------------------------
extern "C" void kernel_launch(void* const* d_in, const int* in_sizes, int n_in,
                              void* d_out, int out_size, void* d_ws, size_t ws_size,
                              hipStream_t stream)
{
  (void)in_sizes; (void)n_in; (void)out_size; (void)ws_size;
  const float* x      = (const float*)d_in[0];
  const float* q_w    = (const float*)d_in[1];
  const float* q_b    = (const float*)d_in[2];
  const float* k_w    = (const float*)d_in[3];
  const float* k_b    = (const float*)d_in[4];
  const float* v_w    = (const float*)d_in[5];
  const float* v_b    = (const float*)d_in[6];
  const float* out_w  = (const float*)d_in[7];
  const float* out_b  = (const float*)d_in[8];
  const float* fuse_w = (const float*)d_in[9];
  const float* fuse_b = (const float*)d_in[10];

  char* ws = (char*)d_ws;
  unsigned short* sw = (unsigned short*)ws;           // 4 swizzled bf16 mats, 131072 B
  float* w3_tmp = (float*)(ws + 131072);              // 65536 B
  float* c2     = (float*)(ws + 196608);              // 512 B
  float* btot   = (float*)(ws + 197120);              // 512 B
  float* stats  = (float*)(ws + 197632);              // 32 B
  float* aff    = (float*)(ws + 262144);              // 1 MB

  prep_w3_kernel<<<128, 128, 0, stream>>>(v_w, v_b, out_w, out_b, fuse_w, fuse_b,
                                          w3_tmp, c2, btot);
  prep_swz_kernel<<<256, 256, 0, stream>>>(q_w, k_w, fuse_w, w3_tmp, sw);
  aff_kernel<<<4096, 128, 0, stream>>>(x, sw, q_b, k_b, aff);
  softmax_stats_kernel<<<4, 256, 0, stream>>>(aff, stats);
  out_kernel<<<4096, 128, 0, stream>>>(x, sw + 2 * WMAT, c2, btot, aff, stats,
                                       (float*)d_out);
}
